// Model_64269890618068
// MI455X (gfx1250) — compile-verified, matrix operand order checked by reading, not run on hardware
//
#include <hip/hip_runtime.h>
#include <hip/hip_bf16.h>

// ---------------------------------------------------------------------------
// BiLSTM-CRF on MI455X (gfx1250): bf16 WMMA GEMMs + register-resident LSTM
// cell state + atomic grid barrier for the recurrent dependence.
// Re-tiled (8 N-tiles/wave, launch_bounds) to eliminate accumulator spills.
// ---------------------------------------------------------------------------

#define T_LEN 256
#define B_N   128
#define E_DIM 256
#define H_DIM 256
#define HD_DIM 512
#define C_N   9

typedef __attribute__((ext_vector_type(16))) __bf16 v16bf;
typedef __attribute__((ext_vector_type(8)))  float  v8f;

union FragU {
    unsigned int u[8];
    v16bf v;
};

__device__ __forceinline__ unsigned short f2bf(float f) {
    unsigned int u = __float_as_uint(f);
    u += 0x7FFFu + ((u >> 16) & 1u);        // round-to-nearest-even
    return (unsigned short)(u >> 16);
}
__device__ __forceinline__ float bf2f(unsigned short h) {
    return __uint_as_float(((unsigned int)h) << 16);
}
__device__ __forceinline__ float sigm(float x) { return 1.0f / (1.0f + __expf(-x)); }

// Load a 16x32 bf16 fragment (A layout; B uses the same packing with the lane
// indexing N instead of M) from row-major storage with leading dim `ld`.
// lane&15 -> row, lane>>4 selects K-half {0,8}; VGPR j packs K pairs.
__device__ __forceinline__ void load_frag(FragU& f, const unsigned short* base, int ld) {
    int lane = threadIdx.x & 31;
    int r    = lane & 15;
    int kh   = (lane >> 4) * 8;
    const unsigned short* p = base + r * ld + kh;
#pragma unroll
    for (int j = 0; j < 4; ++j) f.u[j]     = *(const unsigned int*)(p + 2 * j);
    p += 16;
#pragma unroll
    for (int j = 0; j < 4; ++j) f.u[4 + j] = *(const unsigned int*)(p + 2 * j);
}

// ------------------------------ conversions --------------------------------

__global__ void k_f32_to_bf16(const float* __restrict__ src,
                              unsigned short* __restrict__ dst, int n) {
    int i = blockIdx.x * blockDim.x + threadIdx.x;
    if (i < n) dst[i] = f2bf(src[i]);
}

// Embedding gather -> time-major bf16 [T, B, E]
__global__ void k_embed(const int* __restrict__ x, const float* __restrict__ emb,
                        unsigned short* __restrict__ out) {
    int idx = blockIdx.x * blockDim.x + threadIdx.x;   // ((t*B)+b)*E + e
    int e  = idx & (E_DIM - 1);
    int tb = idx >> 8;                                  // E_DIM == 256
    int t  = tb / B_N;
    int b  = tb - t * B_N;
    int tok = x[b * T_LEN + t];
    out[idx] = f2bf(emb[(size_t)tok * E_DIM + e]);
}

// ------------------------------ batched GEMM -------------------------------
// out[M,1024] (f32) = A[M,K](bf16) @ W[1024,K]^T(bf16) + b1 + b2
// block = 256 thr = 8 waves; wave -> 16 M-rows, 8 N-tiles (128 cols).
__global__ void __launch_bounds__(256, 1)
k_gemm_proj(const unsigned short* __restrict__ A, int lda,
            const unsigned short* __restrict__ W, int ldw,
            const float* __restrict__ b1, const float* __restrict__ b2,
            float* __restrict__ out, int K) {
    int wave = threadIdx.x >> 5;
    int lane = threadIdx.x & 31;
    int m0 = blockIdx.x * 128 + wave * 16;
    int n0 = blockIdx.y * 128;

    v8f acc[8] = {};
    for (int kk = 0; kk < K; kk += 32) {
        FragU a;
        load_frag(a, A + (size_t)m0 * lda + kk, lda);
#pragma unroll
        for (int nt = 0; nt < 8; ++nt) {
            FragU bf;
            load_frag(bf, W + (size_t)(n0 + nt * 16) * ldw + kk, ldw);
            acc[nt] = __builtin_amdgcn_wmma_f32_16x16x32_bf16(
                false, a.v, false, bf.v, (short)0, acc[nt], false, false);
        }
    }
    int n = lane & 15, mo = (lane >> 4) * 8;
#pragma unroll
    for (int nt = 0; nt < 8; ++nt) {
        int col = n0 + nt * 16 + n;
        float bb = b1[col] + b2[col];
#pragma unroll
        for (int r = 0; r < 8; ++r) {
            size_t m = (size_t)m0 + mo + r;
            out[m * 1024 + col] = acc[nt][r] + bb;
        }
    }
}

// ------------------------------ recurrent LSTM -----------------------------
// hbuf: bf16 [(T+2), B, 512]; slot s holds h at time s-1; slot 0 and T+1 are
// zero (initial states for fwd/bwd). Each direction writes its 256-col half,
// so hbuf doubles as the concatenated input of the next layer.
// grid = (8, 2): blockIdx.x -> 32 h-columns, blockIdx.y -> direction.
// Each wave: 16 batch rows x 8 gate-tiles (i/f/g/o x 32 cols); c-state lives
// in registers across all T steps; cross-WG sync via atomic barrier.
__global__ void __launch_bounds__(256, 1)
k_lstm_rec(unsigned short* __restrict__ hbuf,
           const unsigned short* __restrict__ whh_f,
           const unsigned short* __restrict__ whh_b,
           const float* __restrict__ gates_f,
           const float* __restrict__ gates_b,
           unsigned int* __restrict__ cnt) {
    const int dir = blockIdx.y;
    const unsigned short* whh = dir ? whh_b : whh_f;
    const float* gates        = dir ? gates_b : gates_f;
    const int dirofs = dir * H_DIM;
    unsigned int* myCnt = cnt + dir * 32;   // separate cache lines per dir

    const int hs   = blockIdx.x * 32;
    const int wave = threadIdx.x >> 5;
    const int lane = threadIdx.x & 31;
    const int m0   = wave * 16;
    const int n    = lane & 15;
    const int mo   = (lane >> 4) * 8;

    float cst[16];
#pragma unroll
    for (int i = 0; i < 16; ++i) cst[i] = 0.0f;

    for (int t = 0; t < T_LEN; ++t) {
        const int tt        = dir ? (T_LEN - 1 - t) : t;
        const int slot_prev = dir ? (tt + 2) : tt;
        const int slot_cur  = tt + 1;

        const unsigned short* hprev =
            hbuf + (size_t)slot_prev * (B_N * HD_DIM) + dirofs;

        v8f acc[8] = {};
        for (int kk = 0; kk < H_DIM; kk += 32) {
            FragU a;
            load_frag(a, hprev + m0 * HD_DIM + kk, HD_DIM);
#pragma unroll
            for (int nt = 0; nt < 8; ++nt) {
                // gate (nt>>1): columns [gate*H + hs + (nt&1)*16, +16)
                int gcol0 = (nt >> 1) * H_DIM + hs + (nt & 1) * 16;
                FragU bf;
                load_frag(bf, whh + (size_t)gcol0 * H_DIM + kk, H_DIM);
                acc[nt] = __builtin_amdgcn_wmma_f32_16x16x32_bf16(
                    false, a.v, false, bf.v, (short)0, acc[nt], false, false);
            }
        }

        const float* gp = gates + (size_t)tt * (B_N * 1024);
        unsigned short* hcur =
            hbuf + (size_t)slot_cur * (B_N * HD_DIM) + dirofs;

#pragma unroll
        for (int nt0 = 0; nt0 < 2; ++nt0) {
#pragma unroll
            for (int r = 0; r < 8; ++r) {
                int m  = m0 + mo + r;
                int cl = nt0 * 16 + n;          // 0..31 within this block's slice
                size_t rowo = (size_t)m * 1024;
                float gi = acc[nt0][r]     + gp[rowo + 0 * H_DIM + hs + cl];
                float gf = acc[nt0 + 2][r] + gp[rowo + 1 * H_DIM + hs + cl];
                float gg = acc[nt0 + 4][r] + gp[rowo + 2 * H_DIM + hs + cl];
                float go = acc[nt0 + 6][r] + gp[rowo + 3 * H_DIM + hs + cl];
                float ii = sigm(gi), ff = sigm(gf), oo = sigm(go);
                float tg = tanhf(gg);
                int ci = nt0 * 8 + r;
                float cv = ff * cst[ci] + ii * tg;
                cst[ci] = cv;
                hcur[(size_t)m * HD_DIM + hs + cl] = f2bf(oo * tanhf(cv));
            }
        }

        // release/acquire barrier across the 8 blocks of this direction
        __syncthreads();
        if (threadIdx.x == 0) {
            __threadfence();
            atomicAdd(myCnt, 1u);
            unsigned int target = 8u * (unsigned)(t + 1);
            while (atomicAdd(myCnt, 0u) < target) { __builtin_amdgcn_s_sleep(1); }
        }
        __syncthreads();
        __threadfence();
    }
}

// ------------------------------ emission -----------------------------------
// em[T,B,C] = h2 @ lin_w^T + lin_b ; lin_w staged in LDS.
__global__ void k_emission(const unsigned short* __restrict__ h2,
                           const float* __restrict__ lin_w,
                           const float* __restrict__ lin_b,
                           float* __restrict__ em) {
    __shared__ float w_s[C_N * HD_DIM];
    __shared__ float b_s[C_N];
    for (int i = threadIdx.x; i < C_N * HD_DIM; i += blockDim.x) w_s[i] = lin_w[i];
    if (threadIdx.x < C_N) b_s[threadIdx.x] = lin_b[threadIdx.x];
    __syncthreads();

    int idx = blockIdx.x * blockDim.x + threadIdx.x;     // t*B + b
    const unsigned short* hr = h2 + (size_t)idx * HD_DIM;
    float accv[C_N];
#pragma unroll
    for (int c = 0; c < C_N; ++c) accv[c] = b_s[c];
    for (int k = 0; k < HD_DIM; ++k) {
        float hv = bf2f(hr[k]);
#pragma unroll
        for (int c = 0; c < C_N; ++c) accv[c] += hv * w_s[c * HD_DIM + k];
    }
#pragma unroll
    for (int c = 0; c < C_N; ++c) em[(size_t)idx * C_N + c] = accv[c];
}

// ------------------------------ CRF NLL ------------------------------------
// One workgroup, thread b handles batch row b; LDS reduction to scalar.
__global__ void k_crf(const float* __restrict__ em, const int* __restrict__ tags,
                      const unsigned char* __restrict__ mask,
                      const float* __restrict__ trans, const float* __restrict__ start,
                      const float* __restrict__ endv, float* __restrict__ out) {
    __shared__ float red[B_N];
    int b = threadIdx.x;
    float alpha[C_N];
    const float* e0 = em + (size_t)b * C_N;
#pragma unroll
    for (int c = 0; c < C_N; ++c) alpha[c] = start[c] + e0[c];

    int tprev = tags[b * T_LEN + 0];
    float num = start[tprev] + e0[tprev];

    int len = 0;
    for (int t = 0; t < T_LEN; ++t) len += mask[b * T_LEN + t] ? 1 : 0;

    for (int t = 1; t < T_LEN; ++t) {
        const float* e = em + ((size_t)t * B_N + b) * C_N;
        float mm = mask[b * T_LEN + t] ? 1.0f : 0.0f;
        int tc = tags[b * T_LEN + t];
        num += mm * (trans[tprev * C_N + tc] + e[tc]);
        float na[C_N];
#pragma unroll
        for (int c2 = 0; c2 < C_N; ++c2) {
            float mx = -3.4e38f;
#pragma unroll
            for (int c1 = 0; c1 < C_N; ++c1)
                mx = fmaxf(mx, alpha[c1] + trans[c1 * C_N + c2]);
            float s = 0.0f;
#pragma unroll
            for (int c1 = 0; c1 < C_N; ++c1)
                s += __expf(alpha[c1] + trans[c1 * C_N + c2] - mx);
            na[c2] = mx + __logf(s) + e[c2];
        }
        if (mm > 0.0f) {
#pragma unroll
            for (int c = 0; c < C_N; ++c) alpha[c] = na[c];
        }
        tprev = tc;
    }
    int lastTag = tags[b * T_LEN + (len - 1)];
    num += endv[lastTag];

    float mx = -3.4e38f;
#pragma unroll
    for (int c = 0; c < C_N; ++c) mx = fmaxf(mx, alpha[c] + endv[c]);
    float s = 0.0f;
#pragma unroll
    for (int c = 0; c < C_N; ++c) s += __expf(alpha[c] + endv[c] - mx);
    float logZ = mx + __logf(s);

    red[b] = num - logZ;
    __syncthreads();
    for (int st = B_N / 2; st > 0; st >>= 1) {
        if (b < st) red[b] += red[b + st];
        __syncthreads();
    }
    if (b == 0) out[0] = -(red[0] / (float)B_N);
}

// ------------------------------ host ---------------------------------------

extern "C" void kernel_launch(void* const* d_in, const int* in_sizes, int n_in,
                              void* d_out, int out_size, void* d_ws, size_t ws_size,
                              hipStream_t stream) {
    (void)in_sizes; (void)n_in; (void)out_size; (void)ws_size;

    const int*   x     = (const int*)d_in[0];
    const int*   tags  = (const int*)d_in[1];
    const unsigned char* mask = (const unsigned char*)d_in[2];
    const float* emb   = (const float*)d_in[3];
    const float* wih[4] = { (const float*)d_in[4],  (const float*)d_in[8],
                            (const float*)d_in[12], (const float*)d_in[16] };
    const float* whh[4] = { (const float*)d_in[5],  (const float*)d_in[9],
                            (const float*)d_in[13], (const float*)d_in[17] };
    const float* bih[4] = { (const float*)d_in[6],  (const float*)d_in[10],
                            (const float*)d_in[14], (const float*)d_in[18] };
    const float* bhh[4] = { (const float*)d_in[7],  (const float*)d_in[11],
                            (const float*)d_in[15], (const float*)d_in[19] };
    const float* lin_w = (const float*)d_in[20];
    const float* lin_b = (const float*)d_in[21];
    const float* trans = (const float*)d_in[22];
    const float* startv = (const float*)d_in[23];
    const float* endv   = (const float*)d_in[24];
    float* out = (float*)d_out;

    char* ws = (char*)d_ws;
    size_t off = 0;
    auto nxt = [&](size_t bytes) {
        size_t o = off;
        off += (bytes + 255) & ~(size_t)255;
        return o;
    };
    const size_t oXBF   = nxt((size_t)T_LEN * B_N * E_DIM * 2);
    size_t oWIH[4], oWHH[4];
    const int wihK[4] = { E_DIM, E_DIM, HD_DIM, HD_DIM };
    for (int i = 0; i < 4; ++i) oWIH[i] = nxt((size_t)1024 * wihK[i] * 2);
    for (int i = 0; i < 4; ++i) oWHH[i] = nxt((size_t)1024 * H_DIM * 2);
    const size_t hbufElems = (size_t)(T_LEN + 2) * B_N * HD_DIM;
    const size_t oH1  = nxt(hbufElems * 2);
    const size_t oH2  = nxt(hbufElems * 2);
    const size_t oGF  = nxt((size_t)T_LEN * B_N * 1024 * 4);
    const size_t oGB  = nxt((size_t)T_LEN * B_N * 1024 * 4);
    const size_t oEM  = nxt((size_t)T_LEN * B_N * C_N * 4);
    const size_t oCNT = nxt(512);

    unsigned short* XBF = (unsigned short*)(ws + oXBF);
    unsigned short* H1  = (unsigned short*)(ws + oH1);
    unsigned short* H2  = (unsigned short*)(ws + oH2);
    float* GF = (float*)(ws + oGF);
    float* GB = (float*)(ws + oGB);
    float* EM = (float*)(ws + oEM);
    unsigned int* CNT = (unsigned int*)(ws + oCNT);

    const size_t slotBytes = (size_t)B_N * HD_DIM * 2;
    // zero initial-state slots and barrier counters
    hipMemsetAsync(H1, 0, slotBytes, stream);
    hipMemsetAsync((char*)H1 + (size_t)(T_LEN + 1) * slotBytes, 0, slotBytes, stream);
    hipMemsetAsync(H2, 0, slotBytes, stream);
    hipMemsetAsync((char*)H2 + (size_t)(T_LEN + 1) * slotBytes, 0, slotBytes, stream);
    hipMemsetAsync(CNT, 0, 512, stream);

    // weight conversions f32 -> bf16
    for (int i = 0; i < 4; ++i) {
        int n1 = 1024 * wihK[i];
        k_f32_to_bf16<<<(n1 + 255) / 256, 256, 0, stream>>>(
            wih[i], (unsigned short*)(ws + oWIH[i]), n1);
        int n2 = 1024 * H_DIM;
        k_f32_to_bf16<<<(n2 + 255) / 256, 256, 0, stream>>>(
            whh[i], (unsigned short*)(ws + oWHH[i]), n2);
    }

    // embedding gather -> [T,B,E] bf16
    k_embed<<<(T_LEN * B_N * E_DIM) / 256, 256, 0, stream>>>(x, emb, XBF);

    const int M = T_LEN * B_N;
    dim3 gGemm(M / 128, 8);

    // layer 0: input projections (fwd/bwd), then recurrence
    k_gemm_proj<<<gGemm, 256, 0, stream>>>(XBF, E_DIM, (unsigned short*)(ws + oWIH[0]),
                                           E_DIM, bih[0], bhh[0], GF, E_DIM);
    k_gemm_proj<<<gGemm, 256, 0, stream>>>(XBF, E_DIM, (unsigned short*)(ws + oWIH[1]),
                                           E_DIM, bih[1], bhh[1], GB, E_DIM);
    k_lstm_rec<<<dim3(8, 2), 256, 0, stream>>>(
        H1, (unsigned short*)(ws + oWHH[0]), (unsigned short*)(ws + oWHH[1]),
        GF, GB, CNT);

    // layer 1: input is H1 slots 1..T ([T,B,512] bf16, concat of both dirs)
    const unsigned short* H1seq = H1 + (size_t)B_N * HD_DIM;
    k_gemm_proj<<<gGemm, 256, 0, stream>>>(H1seq, HD_DIM, (unsigned short*)(ws + oWIH[2]),
                                           HD_DIM, bih[2], bhh[2], GF, HD_DIM);
    k_gemm_proj<<<gGemm, 256, 0, stream>>>(H1seq, HD_DIM, (unsigned short*)(ws + oWIH[3]),
                                           HD_DIM, bih[3], bhh[3], GB, HD_DIM);
    k_lstm_rec<<<dim3(8, 2), 256, 0, stream>>>(
        H2, (unsigned short*)(ws + oWHH[2]), (unsigned short*)(ws + oWHH[3]),
        GF, GB, CNT + 64);

    // emissions + CRF
    const unsigned short* H2seq = H2 + (size_t)B_N * HD_DIM;
    k_emission<<<(T_LEN * B_N) / 256, 256, 0, stream>>>(H2seq, lin_w, lin_b, EM);
    k_crf<<<1, B_N, 0, stream>>>(EM, tags, mask, trans, startv, endv, out);
}
